// BotImpact_65979287601567
// MI455X (gfx1250) — matrix operand order repaired and checked by reading, not compiled, not run on hardware
//
#include <hip/hip_runtime.h>

typedef __attribute__((ext_vector_type(16))) _Float16 v16h;
typedef __attribute__((ext_vector_type(8)))  float    v8f;

#define NN    50000
#define EE    1600000
#define ENL   (EE + NN)          // edges + self loops
#define CH    64
#define TROWS 25000

// ---- float <-> monotonic-ordered uint (for atomic float max incl. negatives) ----
__device__ __forceinline__ unsigned ordf(float f) {
  unsigned u = __float_as_uint(f);
  return (u & 0x80000000u) ? ~u : (u | 0x80000000u);
}
__device__ __forceinline__ float unordf(unsigned u) {
  return __uint_as_float((u & 0x80000000u) ? (u & 0x7fffffffu) : ~u);
}
#define ORD_NEG_INF 0x007FFFFFu   // ordf(-inf)

__device__ __forceinline__ float leaky(float v, float s) { return v > 0.f ? v : s * v; }

// =====================================================================
// WMMA GEMM: H[M,64] = X[M,K] @ W[K,64]  (f16 inputs, f32 accumulate)
// Fused epilogue: AS[m] = sum_c H[m,c]*att_s[c], AD likewise.
// One wave per 16-row tile; 8 waves / block.
// W is staged to LDS *pre-packed in B-fragment order* so each lane's
// 16 halves are contiguous (one 32B LDS read -> 2x ds_load_b128).
//   packed index: (((kk*4 + nt)*2 + hi)*16 + lm)*16 + e
//   holds        W[(kk*32 + hi*8 + (e&7) + (e>>3)*16)*64 + nt*16 + lm]
// All four N-tile fragments are loaded into registers before the WMMA
// chain so the four v_wmma issue back-to-back behind one dscnt wait.
// =====================================================================
template <int K>
__global__ __launch_bounds__(256)
void gemm_att_kernel(const float* __restrict__ X, const float* __restrict__ W,
                     const float* __restrict__ att_s, const float* __restrict__ att_d,
                     float* __restrict__ H, float* __restrict__ AS, float* __restrict__ AD,
                     int mtiles) {
  __shared__ __align__(32) _Float16 Wl[K * CH];
  for (int i = threadIdx.x; i < K * CH; i += 256) {
    const int e  = i & 15;
    const int lm = (i >> 4) & 15;
    const int hi = (i >> 8) & 1;
    const int nt = (i >> 9) & 3;
    const int kk = i >> 11;
    Wl[i] = (_Float16)W[(kk * 32 + hi * 8 + (e & 7) + (e >> 3) * 16) * CH + nt * 16 + lm];
  }
  __syncthreads();

  const int wave = threadIdx.x >> 5;
  const int lane = threadIdx.x & 31;
  const int tile = blockIdx.x * 8 + wave;
  if (tile >= mtiles) return;

  const int lm = lane & 15;    // M (A/C) or N (B/C) sub-index
  const int hi = lane >> 4;    // half-wave selector

  v8f acc[4] = {v8f{}, v8f{}, v8f{}, v8f{}};
  const float* xrow = X + (size_t)(tile * 16 + lm) * K;
  // per-lane packed-B base (halves): (((kk*4+nt)*2+hi)*16+lm)*16
  const int bbase = (hi * 16 + lm) * 16;

#pragma unroll
  for (int kk = 0; kk < K / 32; ++kk) {
    const int k0 = kk * 32;
    // ---- A fragment (16x32 f16). lane<16: halves0-7 =K0-7, 8-15=K16-23;
    //      lane>=16: halves0-7 =K8-15, 8-15=K24-31.
    const float* xp = xrow + k0 + hi * 8;
    float4 f0 = *(const float4*)(xp);
    float4 f1 = *(const float4*)(xp + 4);
    float4 f2 = *(const float4*)(xp + 16);
    float4 f3 = *(const float4*)(xp + 20);

    // ---- all four B fragments up front (4 x 32B contiguous LDS reads)
    v16h bfrag[4];
#pragma unroll
    for (int nt = 0; nt < 4; ++nt)
      bfrag[nt] = *(const v16h*)(&Wl[((kk * 4 + nt) * 2) * 256 + bbase]);

    v16h a;
    a[0]=(_Float16)f0.x;  a[1]=(_Float16)f0.y;  a[2]=(_Float16)f0.z;  a[3]=(_Float16)f0.w;
    a[4]=(_Float16)f1.x;  a[5]=(_Float16)f1.y;  a[6]=(_Float16)f1.z;  a[7]=(_Float16)f1.w;
    a[8]=(_Float16)f2.x;  a[9]=(_Float16)f2.y;  a[10]=(_Float16)f2.z; a[11]=(_Float16)f2.w;
    a[12]=(_Float16)f3.x; a[13]=(_Float16)f3.y; a[14]=(_Float16)f3.z; a[15]=(_Float16)f3.w;

#pragma unroll
    for (int nt = 0; nt < 4; ++nt)
      acc[nt] = __builtin_amdgcn_wmma_f32_16x16x32_f16(
          false, a, false, bfrag[nt], (short)0, acc[nt], false, false);
  }

  // ---- epilogue: attention dot products + H store ----
  float satt[4], datt[4];
#pragma unroll
  for (int nt = 0; nt < 4; ++nt) {
    satt[nt] = att_s[nt * 16 + lm];
    datt[nt] = att_d[nt * 16 + lm];
  }
  const int mbase = tile * 16 + hi * 8;  // lane<16: rows 0-7; lane>=16: rows 8-15
  float asr[8], adr[8];
#pragma unroll
  for (int r = 0; r < 8; ++r) {
    float ps = 0.f, pd = 0.f;
#pragma unroll
    for (int nt = 0; nt < 4; ++nt) { ps += acc[nt][r] * satt[nt]; pd += acc[nt][r] * datt[nt]; }
    // reduce across the 16 lanes that share this row set
    for (int msk = 1; msk < 16; msk <<= 1) {
      ps += __shfl_xor(ps, msk, 32);
      pd += __shfl_xor(pd, msk, 32);
    }
    asr[r] = ps; adr[r] = pd;
  }
  if (lm == 0) {
#pragma unroll
    for (int r = 0; r < 8; ++r) { AS[mbase + r] = asr[r]; AD[mbase + r] = adr[r]; }
  }
#pragma unroll
  for (int nt = 0; nt < 4; ++nt)
#pragma unroll
    for (int r = 0; r < 8; ++r)
      H[(size_t)(mbase + r) * CH + nt * 16 + lm] = acc[nt][r];
}

// =====================================================================
// Edge-phase kernels (self loops are edge ids >= EE)
// =====================================================================
__global__ void init_conv_kernel(float* __restrict__ agg, unsigned* __restrict__ mord,
                                 float* __restrict__ ssum) {
  int i = blockIdx.x * 256 + threadIdx.x;
  if (i >= NN * CH) return;
  agg[i] = 0.f;
  if (i < NN) { mord[i] = ORD_NEG_INF; ssum[i] = 0.f; }
}

__global__ void edge_max_kernel(const int* __restrict__ ei, const float* __restrict__ AS,
                                const float* __restrict__ AD, float* __restrict__ ebuf,
                                unsigned* __restrict__ mord) {
  int i = blockIdx.x * 256 + threadIdx.x;
  if (i >= ENL) return;
  int s, d;
  if (i < EE) { s = ei[i]; d = ei[EE + i]; } else { s = d = i - EE; }
  float e = leaky(AS[s] + AD[d], 0.2f);
  ebuf[i] = e;
  atomicMax(&mord[d], ordf(e));
}

__global__ void edge_exp_kernel(const int* __restrict__ ei, float* __restrict__ ebuf,
                                const unsigned* __restrict__ mord, float* __restrict__ ssum) {
  int i = blockIdx.x * 256 + threadIdx.x;
  if (i >= ENL) return;
  int d = (i < EE) ? ei[EE + i] : (i - EE);
  float ee = __expf(ebuf[i] - unordf(mord[d]));
  ebuf[i] = ee;
  atomicAdd(&ssum[d], ee);
}

// 16 threads per edge, 4 channels each (float4 gather, scalar atomic scatter)
__global__ void edge_agg_kernel(const int* __restrict__ ei, const float* __restrict__ ebuf,
                                const float* __restrict__ ssum, const float* __restrict__ H,
                                float* __restrict__ agg) {
  int e = blockIdx.x * 16 + (threadIdx.x >> 4);
  if (e >= ENL) return;
  int c = (threadIdx.x & 15) * 4;
  int s, d;
  if (e < EE) { s = ei[e]; d = ei[EE + e]; } else { s = d = e - EE; }
  float alpha = ebuf[e] / ssum[d];
  float4 hv = *(const float4*)(H + (size_t)s * CH + c);
  float* ap = agg + (size_t)d * CH + c;
  atomicAdd(ap + 0, hv.x * alpha);
  atomicAdd(ap + 1, hv.y * alpha);
  atomicAdd(ap + 2, hv.z * alpha);
  atomicAdd(ap + 3, hv.w * alpha);
}

__global__ void finalize_kernel(const float* __restrict__ agg, const float* __restrict__ bias,
                                float* __restrict__ out, int relu) {
  int i = blockIdx.x * 256 + threadIdx.x;
  if (i >= NN * CH) return;
  float v = agg[i] + bias[i & (CH - 1)];
  if (relu) v = fmaxf(v, 0.f);
  out[i] = v;
}

// =====================================================================
// Heads
// =====================================================================
__global__ void tprob_kernel(const float* __restrict__ Z, const float* __restrict__ Wp,
                             const float* __restrict__ bp, float* __restrict__ out) {
  int n = blockIdx.x * 256 + threadIdx.x;
  if (n >= NN) return;
  const float* z = Z + (size_t)n * CH;
  float s0 = bp[0], s1 = bp[1];
#pragma unroll 8
  for (int k = 0; k < CH; ++k) { float zv = z[k]; s0 += zv * Wp[2 * k]; s1 += zv * Wp[2 * k + 1]; }
  out[2 * n] = s0;
  out[2 * n + 1] = s1;
}

// fused: y = leaky( leaky( leaky(z@WyS+byS) @ Wa + ba ) @ Wb + bb ), one wave per row
__global__ __launch_bounds__(256)
void head_kernel(const float* __restrict__ Z, const int* __restrict__ idx, int nrows,
                 const float* __restrict__ WyS, const float* __restrict__ byS,
                 const float* __restrict__ Wa, const float* __restrict__ ba,
                 const float* __restrict__ Wb, const float* __restrict__ bb,
                 float* __restrict__ out) {
  __shared__ float Ws_l[CH * CH];
  __shared__ float Wa_l[CH * CH];
  __shared__ float Wb_l[CH];
  __shared__ float bS_l[CH];
  __shared__ float ba_l[CH];
  for (int i = threadIdx.x; i < CH * CH; i += 256) { Ws_l[i] = WyS[i]; Wa_l[i] = Wa[i]; }
  if (threadIdx.x < CH) {
    Wb_l[threadIdx.x] = Wb[threadIdx.x];
    bS_l[threadIdx.x] = byS[threadIdx.x];
    ba_l[threadIdx.x] = ba[threadIdx.x];
  }
  __syncthreads();

  const int wave = threadIdx.x >> 5;
  const int lane = threadIdx.x & 31;
  const int row = blockIdx.x * 8 + wave;
  if (row >= nrows) return;

  const float* z = Z + (size_t)idx[row] * CH;
  float z0 = z[lane], z1 = z[lane + 32];

  // layer S: 64 -> 64
  float s0 = bS_l[lane], s1 = bS_l[lane + 32];
#pragma unroll 8
  for (int k = 0; k < 32; ++k) {
    float zk = __shfl(z0, k, 32);
    s0 += zk * Ws_l[k * CH + lane];
    s1 += zk * Ws_l[k * CH + lane + 32];
  }
#pragma unroll 8
  for (int k = 0; k < 32; ++k) {
    float zk = __shfl(z1, k, 32);
    s0 += zk * Ws_l[(k + 32) * CH + lane];
    s1 += zk * Ws_l[(k + 32) * CH + lane + 32];
  }
  s0 = leaky(s0, 0.01f); s1 = leaky(s1, 0.01f);

  // layer A: 64 -> 64
  float a0 = ba_l[lane], a1 = ba_l[lane + 32];
#pragma unroll 8
  for (int k = 0; k < 32; ++k) {
    float sk = __shfl(s0, k, 32);
    a0 += sk * Wa_l[k * CH + lane];
    a1 += sk * Wa_l[k * CH + lane + 32];
  }
#pragma unroll 8
  for (int k = 0; k < 32; ++k) {
    float sk = __shfl(s1, k, 32);
    a0 += sk * Wa_l[(k + 32) * CH + lane];
    a1 += sk * Wa_l[(k + 32) * CH + lane + 32];
  }
  a0 = leaky(a0, 0.01f); a1 = leaky(a1, 0.01f);

  // layer B: 64 -> 1
  float y = a0 * Wb_l[lane] + a1 * Wb_l[lane + 32];
  for (int msk = 1; msk < 32; msk <<= 1) y += __shfl_xor(y, msk, 32);
  if (lane == 0) out[row] = leaky(y + bb[0], 0.01f);
}

// =====================================================================
// Host-side launch
// =====================================================================
extern "C" void kernel_launch(void* const* d_in, const int* in_sizes, int n_in,
                              void* d_out, int out_size, void* d_ws, size_t ws_size,
                              hipStream_t stream) {
  (void)in_sizes; (void)n_in; (void)out_size; (void)ws_size;

  const float* x    = (const float*)d_in[0];
  const int*   ei   = (const int*)d_in[1];
  const float* fx   = (const float*)d_in[2];
  const int*   fei  = (const int*)d_in[3];
  const int*   tidx = (const int*)d_in[4];
  const int*   cidx = (const int*)d_in[5];
  const float* W1 = (const float*)d_in[6],  *as1 = (const float*)d_in[7];
  const float* ad1 = (const float*)d_in[8], *b1  = (const float*)d_in[9];
  const float* W2 = (const float*)d_in[10], *as2 = (const float*)d_in[11];
  const float* ad2 = (const float*)d_in[12],*b2  = (const float*)d_in[13];
  const float* WyS = (const float*)d_in[14],*byS = (const float*)d_in[15];
  const float* Wy1a= (const float*)d_in[16],*by1a= (const float*)d_in[17];
  const float* Wy1b= (const float*)d_in[18],*by1b= (const float*)d_in[19];
  const float* Wy0a= (const float*)d_in[20],*by0a= (const float*)d_in[21];
  const float* Wy0b= (const float*)d_in[22],*by0b= (const float*)d_in[23];
  const float* Wp  = (const float*)d_in[24],*bp  = (const float*)d_in[25];

  float* out  = (float*)d_out;
  float* y1   = out;
  float* yc0  = out + TROWS;
  float* y0   = out + 2 * TROWS;
  float* yc1  = out + 3 * TROWS;
  float* xZ2  = out + 4 * TROWS;
  float* xfZ2 = xZ2 + (size_t)NN * CH;
  float* tp   = xfZ2 + (size_t)NN * CH;

  // workspace carve (all 16B aligned)
  float* ws  = (float*)d_ws;
  float* h   = ws;  ws += (size_t)NN * CH;
  float* agg = ws;  ws += (size_t)NN * CH;
  float* z1  = ws;  ws += (size_t)NN * CH;
  float* ebuf= ws;  ws += ENL;
  float* AS  = ws;  ws += NN;
  float* AD  = ws;  ws += NN;
  float* SS  = ws;  ws += NN;
  unsigned* MO = (unsigned*)ws;

  const int MT = NN / 16;                 // 3125 tiles (exact)
  dim3 blk(256);
  dim3 gGemm((MT + 7) / 8);
  dim3 gNC((NN * CH + 255) / 256);
  dim3 gE((ENL + 255) / 256);
  dim3 gAgg((ENL + 15) / 16);
  dim3 gN((NN + 255) / 256);
  dim3 gHead((TROWS + 7) / 8);

  for (int g = 0; g < 2; ++g) {
    const float* X  = g ? fx : x;
    const int*   EI = g ? fei : ei;
    float*       Z2 = g ? xfZ2 : xZ2;
    // ---- conv1: K=128, ReLU output -> z1 ----
    gemm_att_kernel<128><<<gGemm, blk, 0, stream>>>(X, W1, as1, ad1, h, AS, AD, MT);
    init_conv_kernel<<<gNC, blk, 0, stream>>>(agg, MO, SS);
    edge_max_kernel<<<gE, blk, 0, stream>>>(EI, AS, AD, ebuf, MO);
    edge_exp_kernel<<<gE, blk, 0, stream>>>(EI, ebuf, MO, SS);
    edge_agg_kernel<<<gAgg, blk, 0, stream>>>(EI, ebuf, SS, h, agg);
    finalize_kernel<<<gNC, blk, 0, stream>>>(agg, b1, z1, 1);
    // ---- conv2: K=64, linear output -> d_out region ----
    gemm_att_kernel<64><<<gGemm, blk, 0, stream>>>(z1, W2, as2, ad2, h, AS, AD, MT);
    init_conv_kernel<<<gNC, blk, 0, stream>>>(agg, MO, SS);
    edge_max_kernel<<<gE, blk, 0, stream>>>(EI, AS, AD, ebuf, MO);
    edge_exp_kernel<<<gE, blk, 0, stream>>>(EI, ebuf, MO, SS);
    edge_agg_kernel<<<gAgg, blk, 0, stream>>>(EI, ebuf, SS, h, agg);
    finalize_kernel<<<gNC, blk, 0, stream>>>(agg, b2, Z2, 0);
  }

  // ---- heads ----
  tprob_kernel<<<gN, blk, 0, stream>>>(xZ2, Wp, bp, tp);
  head_kernel<<<gHead, blk, 0, stream>>>(xZ2,  tidx, TROWS, WyS, byS, Wy1a, by1a, Wy1b, by1b, y1);
  head_kernel<<<gHead, blk, 0, stream>>>(xfZ2, tidx, TROWS, WyS, byS, Wy0a, by0a, Wy0b, by0b, yc0);
  head_kernel<<<gHead, blk, 0, stream>>>(xZ2,  cidx, TROWS, WyS, byS, Wy0a, by0a, Wy0b, by0b, y0);
  head_kernel<<<gHead, blk, 0, stream>>>(xfZ2, cidx, TROWS, WyS, byS, Wy1a, by1a, Wy1b, by1b, yc1);
}
